// Attention_90211493085859
// MI455X (gfx1250) — matrix operand order repaired and checked by reading, not compiled, not run on hardware
//
#include <hip/hip_runtime.h>
#include <hip/hip_bf16.h>

// ---------------------------------------------------------------------------
// Types for CDNA5 WMMA (wave32): A/B = 16 bf16 per lane (8 VGPRs), C/D = 8 f32
// ---------------------------------------------------------------------------
typedef __attribute__((ext_vector_type(16))) __bf16          v16bf;
typedef __attribute__((ext_vector_type(8)))  float           v8f;
typedef __attribute__((ext_vector_type(8)))  unsigned short  v8u;
typedef __attribute__((ext_vector_type(16))) unsigned short  v16u;

union Frag16 { v16u v; v8u h[2]; };

__device__ __forceinline__ unsigned short f2bf(float f) {
    unsigned int u = __builtin_bit_cast(unsigned int, f);
    u += 0x7FFFu + ((u >> 16) & 1u);          // round-to-nearest-even
    return (unsigned short)(u >> 16);
}

__device__ __forceinline__ v8f wmma_bf16(v16u a, v16u b, v8f c) {
    return __builtin_amdgcn_wmma_f32_16x16x32_bf16(
        /*neg_a=*/false, __builtin_bit_cast(v16bf, a),
        /*neg_b=*/false, __builtin_bit_cast(v16bf, b),
        /*c_mod=*/(short)0, c, /*reuse_a=*/false, /*reuse_b=*/false);
}

// ---------------------------------------------------------------------------
// CDNA5 async global->LDS copy (ASYNCcnt-tracked), with safe fallback.
// Builtin signature (from hipcc diagnostic): (global v4i*, local v4i*, imm, imm)
// ---------------------------------------------------------------------------
#if __has_builtin(__builtin_amdgcn_global_load_async_to_lds_b128)
#define ASYNC_LDS_COPY 1
typedef int gv4i __attribute__((vector_size(16)));
typedef __attribute__((address_space(1))) gv4i as1_v4i;
typedef __attribute__((address_space(3))) gv4i as3_v4i;
#endif

__device__ __forceinline__ void copy16_g2l(const void* g, void* l) {
#ifdef ASYNC_LDS_COPY
    __builtin_amdgcn_global_load_async_to_lds_b128(
        (as1_v4i*)const_cast<void*>(g), (as3_v4i*)l, 0, 0);
#else
    *(v8u*)l = *(const v8u*)g;
#endif
}

__device__ __forceinline__ void wait_g2l() {
#ifdef ASYNC_LDS_COPY
#if __has_builtin(__builtin_amdgcn_s_wait_asynccnt)
    __builtin_amdgcn_s_wait_asynccnt(0);
#else
    asm volatile("s_wait_asynccnt 0x0" ::: "memory");
#endif
#endif
}

// Problem constants
#define BB 2
#define TT 2048
#define CC 2048
#define HH 16
#define DD 128

// ---------------------------------------------------------------------------
// Kernel 1: f32 -> bf16 conversion (vectorized, grid-stride over 8-elem chunks)
// ---------------------------------------------------------------------------
__global__ void cvt_f32_bf16(const float* __restrict__ in,
                             unsigned short* __restrict__ out, int n8) {
    int i = blockIdx.x * blockDim.x + threadIdx.x;
    int stride = gridDim.x * blockDim.x;
    for (; i < n8; i += stride) {
        const float* p = in + (size_t)i * 8;
        v8u o;
        #pragma unroll
        for (int e = 0; e < 8; ++e) o[e] = f2bf(p[e]);
        *(v8u*)(out + (size_t)i * 8) = o;
    }
}

// ---------------------------------------------------------------------------
// Kernel 2: bf16 WMMA GEMM  C[M,N] = A[M,K] * B[K,N]  (f32 out)
//   block = 256 threads = 8 waves; tile 128x128; BK = 32, double-buffered LDS
//   wave grid 4(M) x 2(N): each wave computes 32x64 = 2x4 WMMA tiles
// ---------------------------------------------------------------------------
__global__ __launch_bounds__(256) void gemm_bf16_wmma(
        const unsigned short* __restrict__ A,
        const unsigned short* __restrict__ B,
        float* __restrict__ C, int M, int N, int K) {
    __shared__ __attribute__((aligned(16))) unsigned short sA[2][128 * 32];
    __shared__ __attribute__((aligned(16))) unsigned short sB[2][32 * 128];

    const int tid  = threadIdx.x;
    const int lane = tid & 31;
    const int w    = tid >> 5;
    const int wm   = w & 3;                 // 0..3  -> 32-row strip
    const int wn   = w >> 2;                // 0..1  -> 64-col strip
    const int tileM = blockIdx.y * 128;
    const int tileN = blockIdx.x * 128;

    // Per-thread staging coordinates (2 x 16B chunks each for A and B)
    const int e0 = tid * 8, e1 = (tid + 256) * 8;
    const int ra0 = e0 >> 5,  ca0 = e0 & 31;
    const int ra1 = e1 >> 5,  ca1 = e1 & 31;
    const int rb0 = e0 >> 7,  cb0 = e0 & 127;
    const int rb1 = e1 >> 7,  cb1 = e1 & 127;

    auto stage = [&](int buf, int k0) {
        copy16_g2l(&A[(size_t)(tileM + ra0) * K + k0 + ca0], &sA[buf][e0]);
        copy16_g2l(&A[(size_t)(tileM + ra1) * K + k0 + ca1], &sA[buf][e1]);
        copy16_g2l(&B[(size_t)(k0 + rb0) * N + tileN + cb0], &sB[buf][e0]);
        copy16_g2l(&B[(size_t)(k0 + rb1) * N + tileN + cb1], &sB[buf][e1]);
    };

    v8f acc[2][4];
    for (int mi = 0; mi < 2; ++mi)
        for (int ni = 0; ni < 4; ++ni) acc[mi][ni] = (v8f){};

    const int aRow  = lane & 15;            // M within 16-row tile
    const int kHalf = (lane >> 4) * 8;      // A-frag K base per half-wave

    stage(0, 0);
    int buf = 0;
    for (int k0 = 0; k0 < K; k0 += 32) {
        wait_g2l();                          // my async stage landed in LDS
        __syncthreads();                     // everyone's stage landed
        if (k0 + 32 < K) stage(buf ^ 1, k0 + 32);   // overlap next stage

        // A fragments: lane holds row M=aRow, K = kHalf+0..7 and kHalf+16..23
        v16u afr[2];
        #pragma unroll
        for (int mi = 0; mi < 2; ++mi) {
            int m = wm * 32 + mi * 16 + aRow;
            Frag16 u;
            u.h[0] = *(const v8u*)&sA[buf][m * 32 + kHalf];
            u.h[1] = *(const v8u*)&sA[buf][m * 32 + kHalf + 16];
            afr[mi] = u.v;
        }
        // B fragments: lane holds column N=nb+aRow, K = kHalf+0..7 and kHalf+16..23
        v16u bfr[4];
        #pragma unroll
        for (int ni = 0; ni < 4; ++ni) {
            int nb = wn * 64 + ni * 16;
            Frag16 u;
            #pragma unroll
            for (int s = 0; s < 16; ++s)
                u.v[s] = sB[buf][((s < 8) ? (kHalf + s) : (kHalf + 8 + s)) * 128 + nb + aRow];
            bfr[ni] = u.v;
        }
        #pragma unroll
        for (int mi = 0; mi < 2; ++mi)
            #pragma unroll
            for (int ni = 0; ni < 4; ++ni)
                acc[mi][ni] = wmma_bf16(afr[mi], bfr[ni], acc[mi][ni]);
        buf ^= 1;
    }

    // C layout: VGPR r, lanes 0-15 -> M=r, lanes 16-31 -> M=r+8, N = lane&15
    const int rowHalf = (lane >> 4) * 8;
    const int cLane   = lane & 15;
    #pragma unroll
    for (int mi = 0; mi < 2; ++mi)
        #pragma unroll
        for (int ni = 0; ni < 4; ++ni)
            #pragma unroll
            for (int r = 0; r < 8; ++r) {
                int m = tileM + wm * 32 + mi * 16 + r + rowHalf;
                int n = tileN + wn * 64 + ni * 16 + cLane;
                C[(size_t)m * N + n] = acc[mi][ni][r];
            }
}

// ---------------------------------------------------------------------------
// Kernel 3: RoPE on q,k; pack q/k/v f32 -> bf16 in [B,H,T,D] layout
// ---------------------------------------------------------------------------
__global__ __launch_bounds__(256) void rope_pack(
        const float* __restrict__ qkv, const float* __restrict__ cosb,
        const float* __restrict__ sinb,
        unsigned short* __restrict__ qo, unsigned short* __restrict__ ko,
        unsigned short* __restrict__ vo) {
    int idx = blockIdx.x * blockDim.x + threadIdx.x;   // b,t,h,d flattened
    int d = idx & (DD - 1);
    int h = (idx >> 7) & (HH - 1);
    int t = (idx >> 11) & (TT - 1);
    int b = idx >> 22;
    const float* row = qkv + (size_t)(b * TT + t) * (3 * CC);
    int hd = h * DD + d;
    float qv = row[hd];
    float kv = row[CC + hd];
    float vv = row[2 * CC + hd];
    int pd = (d < 64) ? (hd + 64) : (hd - 64);
    float sgn = (d < 64) ? -1.0f : 1.0f;
    float qp = sgn * row[pd];
    float kp = sgn * row[CC + pd];
    float c = cosb[t * DD + d], s = sinb[t * DD + d];
    size_t o = (size_t)((b * HH + h) * TT + t) * DD + d;
    qo[o] = f2bf(qv * c + qp * s);
    ko[o] = f2bf(kv * c + kp * s);
    vo[o] = f2bf(vv);
}

// ---------------------------------------------------------------------------
// Kernel 4: flash attention, one (b,h,qblock) per workgroup.
//   128 threads = 4 waves, each wave owns 16 query rows (qblock = 64 rows).
//   Key blocks of 32: S(16x32) = 2 WMMA tiles x 4 K-chunks; online softmax
//   with 16-lane shuffle reductions; P re-laid out through per-wave LDS;
//   O(16x128) = 8 WMMA accumulators. V staged via async global->LDS.
// ---------------------------------------------------------------------------
__global__ __launch_bounds__(128) void flash_attn_wmma(
        const unsigned short* __restrict__ Q,
        const unsigned short* __restrict__ Kc,
        const unsigned short* __restrict__ V,
        unsigned short* __restrict__ Y) {
    __shared__ __attribute__((aligned(16))) unsigned short sKt[DD * 32];   // [d][key]
    __shared__ __attribute__((aligned(16))) unsigned short sV[32 * DD];    // [key][d]
    __shared__ __attribute__((aligned(16))) unsigned short sP[4][16 * 32]; // per-wave

    const int tid  = threadIdx.x;
    const int lane = tid & 31;
    const int w    = tid >> 5;
    const int bid  = blockIdx.x;               // b*H*32 + h*32 + qb
    const int qb   = bid & 31;
    const int h    = (bid >> 5) & (HH - 1);
    const int b    = bid >> 9;
    const size_t headOff = (size_t)(b * HH + h) * TT * DD;
    const int q0 = qb * 64 + w * 16;
    const float scale = 0.08838834764831845f;  // 1/sqrt(128)

    // Preload Q fragments (16x128 = 4 A-frags of 16x32)
    const int mloc = lane & 15;
    const int kh   = (lane >> 4) * 8;
    v16u qa[4];
    #pragma unroll
    for (int c = 0; c < 4; ++c) {
        const unsigned short* qp = Q + headOff + (size_t)(q0 + mloc) * DD + c * 32 + kh;
        Frag16 u;
        u.h[0] = *(const v8u*)qp;
        u.h[1] = *(const v8u*)(qp + 16);
        qa[c] = u.v;
    }

    v8f O[8];
    #pragma unroll
    for (int j = 0; j < 8; ++j) O[j] = (v8f){};
    float mrow[8], lrow[8];
    #pragma unroll
    for (int r = 0; r < 8; ++r) { mrow[r] = -3.0e38f; lrow[r] = 0.0f; }

    for (int kb = 0; kb < TT; kb += 32) {
        __syncthreads();
        {   // V block 32x128 -> row-major LDS (async path)
            #pragma unroll
            for (int i = 0; i < 4; ++i) {
                int e = (tid + i * 128) * 8;
                int rr = e >> 7, cc = e & 127;
                copy16_g2l(&V[headOff + (size_t)(kb + rr) * DD + cc], &sV[e]);
            }
            // K block 32x128 -> transposed LDS [d][key]
            int key = tid >> 2, db = (tid & 3) * 32;
            const unsigned short* kp = Kc + headOff + (size_t)(kb + key) * DD + db;
            #pragma unroll
            for (int j = 0; j < 4; ++j) {
                v8u kv = *(const v8u*)(kp + j * 8);
                #pragma unroll
                for (int e = 0; e < 8; ++e)
                    sKt[(db + j * 8 + e) * 32 + key] = kv[e];
            }
        }
        wait_g2l();
        __syncthreads();

        // S = Q * K^T  (two 16x16 tiles over 32 keys)
        v8f s0 = (v8f){}, s1 = (v8f){};
        #pragma unroll
        for (int c = 0; c < 4; ++c) {
            Frag16 u0, u1;                     // B-frag: lane holds key column, d in table order
            #pragma unroll
            for (int s = 0; s < 16; ++s) {
                int dr = c * 32 + ((s < 8) ? (kh + s) : (kh + 8 + s));
                u0.v[s] = sKt[dr * 32 + mloc];
                u1.v[s] = sKt[dr * 32 + 16 + mloc];
            }
            s0 = wmma_bf16(qa[c], u0.v, s0);
            s1 = wmma_bf16(qa[c], u1.v, s1);
        }

        // Online softmax per row (rows live in 16-lane groups)
        float p0[8], p1[8], alpha[8];
        #pragma unroll
        for (int r = 0; r < 8; ++r) {
            float v0 = s0[r] * scale, v1 = s1[r] * scale;
            float mx = fmaxf(v0, v1);
            #pragma unroll
            for (int off = 8; off >= 1; off >>= 1)
                mx = fmaxf(mx, __shfl_xor(mx, off, 32));
            float mnew = fmaxf(mrow[r], mx);
            float a  = __expf(mrow[r] - mnew);
            float e0 = __expf(v0 - mnew);
            float e1 = __expf(v1 - mnew);
            float sum = e0 + e1;
            #pragma unroll
            for (int off = 8; off >= 1; off >>= 1)
                sum += __shfl_xor(sum, off, 32);
            lrow[r] = lrow[r] * a + sum;
            mrow[r] = mnew;
            alpha[r] = a; p0[r] = e0; p1[r] = e1;
        }
        #pragma unroll
        for (int j = 0; j < 8; ++j)
            #pragma unroll
            for (int r = 0; r < 8; ++r) O[j][r] *= alpha[r];

        // P (C layout) -> per-wave LDS -> A-frag layout
        {
            int nr = lane & 15, rh = (lane >> 4) * 8;
            #pragma unroll
            for (int r = 0; r < 8; ++r) {
                sP[w][(r + rh) * 32 + nr]      = f2bf(p0[r]);
                sP[w][(r + rh) * 32 + 16 + nr] = f2bf(p1[r]);
            }
        }
        Frag16 pu;
        pu.h[0] = *(const v8u*)&sP[w][mloc * 32 + kh];
        pu.h[1] = *(const v8u*)&sP[w][mloc * 32 + kh + 16];

        // O += P * V  (8 N-tiles over D=128)
        #pragma unroll
        for (int j = 0; j < 8; ++j) {
            Frag16 vu;                          // B-frag: lane holds d column, keys in table order
            #pragma unroll
            for (int s = 0; s < 16; ++s)
                vu.v[s] = sV[((s < 8) ? (kh + s) : (kh + 8 + s)) * DD + j * 16 + mloc];
            O[j] = wmma_bf16(pu.v, vu.v, O[j]);
        }
    }

    // Normalize and write Y in [B,T,C] bf16 layout
    const int rh = (lane >> 4) * 8, nc = lane & 15;
    #pragma unroll
    for (int r = 0; r < 8; ++r) {
        float inv = lrow[r] > 0.0f ? 1.0f / lrow[r] : 0.0f;
        int t = q0 + r + rh;
        size_t base = (size_t)(b * TT + t) * CC + h * DD;
        #pragma unroll
        for (int j = 0; j < 8; ++j)
            Y[base + j * 16 + nc] = f2bf(O[j][r] * inv);
    }
}

// ---------------------------------------------------------------------------
// Host launcher
// ---------------------------------------------------------------------------
extern "C" void kernel_launch(void* const* d_in, const int* in_sizes, int n_in,
                              void* d_out, int out_size, void* d_ws, size_t ws_size,
                              hipStream_t stream) {
    const float* x     = (const float*)d_in[0];  // [B,T,C]
    const float* cosb  = (const float*)d_in[1];  // [T,D]
    const float* sinb  = (const float*)d_in[2];  // [T,D]
    const float* Wqkv  = (const float*)d_in[3];  // [C,3C]
    const float* Wproj = (const float*)d_in[4];  // [C,C]
    float* out = (float*)d_out;

    const int XN    = BB * TT * CC;          // 8388608
    const int WQKVN = CC * 3 * CC;           // 12582912
    const int WPRJN = CC * CC;               // 4194304
    const int QKVN  = BB * TT * 3 * CC;      // 25165824
    const int HEADN = BB * HH * TT * DD;     // 8388608

    char* ws = (char*)d_ws;
    size_t off = 0;
    auto alloc = [&](size_t bytes) {
        void* p = ws + off;
        off += (bytes + 255) & ~(size_t)255;
        return p;
    };
    unsigned short* xb   = (unsigned short*)alloc((size_t)XN * 2);
    unsigned short* wqb  = (unsigned short*)alloc((size_t)WQKVN * 2);
    unsigned short* wpb  = (unsigned short*)alloc((size_t)WPRJN * 2);
    float*          qkv  = (float*)alloc((size_t)QKVN * 4);
    unsigned short* qbf  = (unsigned short*)alloc((size_t)HEADN * 2);
    unsigned short* kbf  = (unsigned short*)alloc((size_t)HEADN * 2);
    unsigned short* vbf  = (unsigned short*)alloc((size_t)HEADN * 2);
    unsigned short* ybf  = (unsigned short*)alloc((size_t)XN * 2);
    if (off > ws_size) return;               // workspace too small: bail safely

    // 1) convert inputs to bf16
    cvt_f32_bf16<<<2048, 256, 0, stream>>>(x, xb, XN / 8);
    cvt_f32_bf16<<<2048, 256, 0, stream>>>(Wqkv, wqb, WQKVN / 8);
    cvt_f32_bf16<<<2048, 256, 0, stream>>>(Wproj, wpb, WPRJN / 8);

    // 2) qkv = x @ W_qkv   (M=4096, K=2048, N=6144)
    gemm_bf16_wmma<<<dim3(3 * CC / 128, BB * TT / 128), 256, 0, stream>>>(
        xb, wqb, qkv, BB * TT, 3 * CC, CC);

    // 3) RoPE + pack q/k/v -> [B,H,T,D] bf16
    rope_pack<<<(BB * TT * HH * DD) / 256, 256, 0, stream>>>(
        qkv, cosb, sinb, qbf, kbf, vbf);

    // 4) flash attention -> y [B,T,C] bf16
    flash_attn_wmma<<<BB * HH * (TT / 64), 128, 0, stream>>>(qbf, kbf, vbf, ybf);

    // 5) out = y @ W_proj  (M=4096, K=2048, N=2048)
    gemm_bf16_wmma<<<dim3(CC / 128, BB * TT / 128), 256, 0, stream>>>(
        ybf, wpb, out, BB * TT, CC, CC);
}
